// CombinedGrAFFChemBERTaModel_43456479101555
// MI455X (gfx1250) — matrix operand order repaired and checked by reading, not compiled, not run on hardware
//
#include <hip/hip_runtime.h>
#include <hip/hip_bf16.h>
#include <math.h>

typedef __bf16 bf16_t;
typedef __attribute__((ext_vector_type(16))) __bf16 v16bf;
typedef __attribute__((ext_vector_type(8)))  float   v8f;

#if defined(__AMDGCN__) && __has_builtin(__builtin_amdgcn_global_load_async_to_lds_b128) && __has_builtin(__builtin_amdgcn_s_wait_asynccnt)
#define ATH_ASYNC 1
#else
#define ATH_ASYNC 0
#endif

#if ATH_ASYNC
typedef int athv4i __attribute__((vector_size(16)));
#define ATH_GPTR(p) ((__attribute__((address_space(1))) athv4i*)(p))
#define ATH_LPTR(p) ((__attribute__((address_space(3))) athv4i*)(p))
#endif

namespace {

constexpr int kB = 1024, kS = 128, kH = 768;
constexpr int kKeig = 8, kPhi = 32, kDepth = 6;
constexpr int kSupp = 256, kFrag = 512;
constexpr int kNA = 10, kNB = 9;
constexpr int kN = kB * kNA;   // 10240 nodes
constexpr int kE = kB * kNB;   // 9216 edges

__device__ __forceinline__ bf16_t f2bf(float f) {
  union { float f; unsigned u; } a; a.f = f;
  unsigned r = a.u + 0x7fffu + ((a.u >> 16) & 1u);   // round-to-nearest-even
  union { unsigned short s; bf16_t b; } o; o.s = (unsigned short)(r >> 16);
  return o.b;
}

__device__ __forceinline__ float bsum(float v, float* red) {
  int t = threadIdx.x;
  red[t] = v; __syncthreads();
  for (int s = blockDim.x >> 1; s > 0; s >>= 1) {
    if (t < s) red[t] += red[t + s];
    __syncthreads();
  }
  float r = red[0]; __syncthreads();
  return r;
}

__device__ __forceinline__ float bmaxr(float v, float* red) {
  int t = threadIdx.x;
  red[t] = v; __syncthreads();
  for (int s = blockDim.x >> 1; s > 0; s >>= 1) {
    if (t < s) red[t] = fmaxf(red[t], red[t + s]);
    __syncthreads();
  }
  float r = red[0]; __syncthreads();
  return r;
}

// ------------------------------------------------------------------
// bf16 WMMA GEMM: C[M,N] = act(A[M,K] @ W[K,N] + bias) [+ resid]
// A[M,K] row-major bf16; Wt = W^T, i.e. [N,K] row-major bf16.
// block = 128 threads (4 wave32). Tile 128x64, K-step 32.
// Each wave: 32 rows x 64 cols = 8x v_wmma_f32_16x16x32_bf16 per K-step.
// LDS double-buffered; staged with GLOBAL_LOAD_ASYNC_TO_LDS_B128 (ASYNCcnt)
// when available, else register staging.
// act: 0 none, 1 relu, 2 silu, 3 gelu(exact)
// ------------------------------------------------------------------
__global__ __launch_bounds__(128) void gemm_bf16(
    const bf16_t* __restrict__ A, const bf16_t* __restrict__ Wt,
    const float* __restrict__ bias, const float* __restrict__ resid,
    float* __restrict__ Cf, bf16_t* __restrict__ Cb,
    int M, int K, int N, int act)
{
  __shared__ __align__(16) bf16_t As[2][128][40];  // rows x K (+8 pad, 80B stride)
  __shared__ __align__(16) bf16_t Bs[2][64][40];   // cols x K (+8 pad)
  const int tid  = threadIdx.x;
  const int lane = tid & 31;
  const int wave = tid >> 5;
  const int half = lane >> 4;
  const int m0 = blockIdx.x * 128;
  const int n0 = blockIdx.y * 64;
  const int ntiles = K >> 5;

  // cooperative staging of one K-tile into LDS buffer `buf`
  auto stage = [&](int t, int buf) {
    const int kk = t << 5;
    #pragma unroll
    for (int j = 0; j < 4; ++j) {                       // A: 512 x b128
      int idx = tid + j * 128, r = idx >> 2, s = idx & 3;
      const bf16_t* gp = A + (size_t)(m0 + r) * K + kk + s * 8;
#if ATH_ASYNC
      __builtin_amdgcn_global_load_async_to_lds_b128(ATH_GPTR(gp),
          ATH_LPTR(&As[buf][r][s * 8]), 0, 0);
#else
      *(uint4*)(&As[buf][r][s * 8]) = *(const uint4*)gp;
#endif
    }
    #pragma unroll
    for (int j = 0; j < 2; ++j) {                       // B: 256 x b128
      int idx = tid + j * 128, r = idx >> 2, s = idx & 3;
      const bf16_t* gp = Wt + (size_t)(n0 + r) * K + kk + s * 8;
#if ATH_ASYNC
      __builtin_amdgcn_global_load_async_to_lds_b128(ATH_GPTR(gp),
          ATH_LPTR(&Bs[buf][r][s * 8]), 0, 0);
#else
      *(uint4*)(&Bs[buf][r][s * 8]) = *(const uint4*)gp;
#endif
    }
  };

  v8f acc[2][4];
  #pragma unroll
  for (int s = 0; s < 2; ++s)
    #pragma unroll
    for (int t = 0; t < 4; ++t)
      #pragma unroll
      for (int i = 0; i < 8; ++i) acc[s][t][i] = 0.0f;

  stage(0, 0);

  for (int t = 0; t < ntiles; ++t) {
    const int cur = t & 1;
    if (t + 1 < ntiles) {
      stage(t + 1, cur ^ 1);            // overlap next tile's copy with math
#if ATH_ASYNC
      __builtin_amdgcn_s_wait_asynccnt(6);   // oldest 6 (tile t) complete
#endif
    } else {
#if ATH_ASYNC
      __builtin_amdgcn_s_wait_asynccnt(0);
#endif
    }
    __syncthreads();                     // tile t visible to all waves

    // A fragments (16x32 bf16): lane -> row (lane&15), K[h8,h8+8) & K[16+h8,..)
    const int ra = (wave << 5) + (lane & 15);
    union { v16bf v; uint4 q[2]; } af0, af1;
    af0.q[0] = *(const uint4*)(&As[cur][ra][half * 8]);
    af0.q[1] = *(const uint4*)(&As[cur][ra][16 + half * 8]);
    af1.q[0] = *(const uint4*)(&As[cur][ra + 16][half * 8]);
    af1.q[1] = *(const uint4*)(&As[cur][ra + 16][16 + half * 8]);

    #pragma unroll
    for (int n = 0; n < 4; ++n) {
      // B fragment (32x16 bf16): lane -> col, K[h8,h8+8) & K[16+h8,..)
      union { v16bf v; uint4 q[2]; } bfm;
      const int c = (n << 4) + (lane & 15);
      bfm.q[0] = *(const uint4*)(&Bs[cur][c][half * 8]);
      bfm.q[1] = *(const uint4*)(&Bs[cur][c][16 + half * 8]);
      acc[0][n] = __builtin_amdgcn_wmma_f32_16x16x32_bf16(
          false, af0.v, false, bfm.v, (short)0, acc[0][n], false, false);
      acc[1][n] = __builtin_amdgcn_wmma_f32_16x16x32_bf16(
          false, af1.v, false, bfm.v, (short)0, acc[1][n], false, false);
    }
    __syncthreads();                     // all reads done before next overwrite
  }

  // C/D layout: VGPR r -> M = r + 8*half, N = lane%16
  const int cbase = n0 + (lane & 15);
  #pragma unroll
  for (int st = 0; st < 2; ++st) {
    const int rbase = m0 + (wave << 5) + (st << 4) + (half << 3);
    #pragma unroll
    for (int n = 0; n < 4; ++n) {
      const int col = cbase + (n << 4);
      const float bv = bias ? bias[col] : 0.0f;
      #pragma unroll
      for (int r = 0; r < 8; ++r) {
        const size_t o = (size_t)(rbase + r) * N + col;
        float v = acc[st][n][r] + bv;
        if (act == 1)      v = fmaxf(v, 0.0f);
        else if (act == 2) v = v / (1.0f + __expf(-v));
        else if (act == 3) v = 0.5f * v * (1.0f + erff(v * 0.70710678f));
        if (resid) v += resid[o];
        if (Cf) Cf[o] = v;
        if (Cb) Cb[o] = f2bf(v);
      }
    }
  }
}

// ---------------- elementwise / glue kernels ----------------

__global__ void cvt_f32_bf16(const float* __restrict__ s, bf16_t* __restrict__ d, int n) {
  int i = blockIdx.x * 256 + threadIdx.x;
  if (i < n) d[i] = f2bf(s[i]);
}

// W[K,N] f32 -> Wt[N,K] bf16, 32x32 LDS-tiled; blockIdx.z = layer slice
__global__ void cvt_transpose(const float* __restrict__ s, bf16_t* __restrict__ d,
                              int K, int N) {
  __shared__ float tile[32][33];
  const size_t zo = (size_t)blockIdx.z * K * N;
  const float* sp = s + zo;
  bf16_t* dp = d + zo;
  int k0 = blockIdx.x * 32, n0 = blockIdx.y * 32;
  int tx = threadIdx.x & 31, ty = threadIdx.x >> 5;   // 32 x 8
  #pragma unroll
  for (int i = ty; i < 32; i += 8)
    tile[i][tx] = sp[(size_t)(k0 + i) * N + n0 + tx];
  __syncthreads();
  #pragma unroll
  for (int i = ty; i < 32; i += 8)
    dp[(size_t)(n0 + i) * K + k0 + tx] = f2bf(tile[tx][i]);
}

// token -> node/edge segment sums (atom t%NA, bond t%NB)
__global__ void build_feats(const float* __restrict__ tok,
                            float* __restrict__ nodef, float* __restrict__ edgef) {
  int i = blockIdx.x * 256 + threadIdx.x;      // over B*H
  if (i >= kB * kH) return;
  int b = i / kH, h = i - b * kH;
  float an[kNA]; float bn[kNB];
  #pragma unroll
  for (int a = 0; a < kNA; ++a) an[a] = 0.0f;
  #pragma unroll
  for (int e = 0; e < kNB; ++e) bn[e] = 0.0f;
  const float* tp = tok + (size_t)b * kS * kH + h;
  #pragma unroll
  for (int t = 0; t < kS; ++t) {
    float v = tp[(size_t)t * kH];
    an[t % kNA] += v;
    bn[t % kNB] += v;
  }
  #pragma unroll
  for (int a = 0; a < kNA; ++a) nodef[(size_t)(b * kNA + a) * kH + h] = an[a];
  #pragma unroll
  for (int e = 0; e < kNB; ++e) edgef[(size_t)(b * kNB + e) * kH + h] = bn[e];
}

// SignNet phi: h[n,k,:] = phi(u) + phi(u*[-1,1])   (2->32 relu, 32->32)
__global__ void signnet_phi(const float* __restrict__ eigvecs, const float* __restrict__ eigvals,
                            const float* __restrict__ W1, const float* __restrict__ b1,
                            const float* __restrict__ W2, const float* __restrict__ b2,
                            bf16_t* __restrict__ hout) {
  int i = blockIdx.x * 128 + threadIdx.x;      // over N*Keig
  if (i >= kN * kKeig) return;
  int n = i / kKeig, k = i - n * kKeig;
  int g = n / kNA;
  float u0 = eigvecs[i];
  float u1 = eigvals[g * kKeig + k];
  float hs[kPhi];
  #pragma unroll
  for (int j = 0; j < kPhi; ++j) {
    float w0 = W1[j], w1 = W1[kPhi + j], bb = b1[j];
    float hp = fmaxf( u0 * w0 + u1 * w1 + bb, 0.0f);
    float hm = fmaxf(-u0 * w0 + u1 * w1 + bb, 0.0f);
    hs[j] = hp + hm;                            // relu per sign, then summed
  }
  bf16_t* o = hout + (size_t)n * (kKeig * kPhi) + k * kPhi;
  for (int d = 0; d < kPhi; ++d) {
    float acc = 2.0f * b2[d];
    #pragma unroll
    for (int j = 0; j < kPhi; ++j) acc += hs[j] * W2[j * kPhi + d];
    o[d] = f2bf(acc);
  }
}

// agg[dst] += relu(x[src] + edge_feat)   (<=1 incoming edge/node in this graph)
__global__ void gine_agg(const float* __restrict__ x, const float* __restrict__ edgef,
                         const int* __restrict__ eidx, float* __restrict__ agg) {
  int i = blockIdx.x * 256 + threadIdx.x;       // over E*H
  if (i >= kE * kH) return;
  int e = i / kH, h = i - e * kH;
  int s = eidx[e], d = eidx[kE + e];
  float v = fmaxf(x[(size_t)s * kH + h] + edgef[i], 0.0f);
  atomicAdd(&agg[(size_t)d * kH + h], v);
}

__global__ void make_hm(const float* __restrict__ x, const float* __restrict__ agg,
                        const float* __restrict__ eps, int l,
                        float* __restrict__ hmf, bf16_t* __restrict__ hmb) {
  int i = blockIdx.x * 256 + threadIdx.x;
  if (i >= kN * kH) return;
  float v = (1.0f + eps[l]) * x[i] + agg[i];
  hmf[i] = v; hmb[i] = f2bf(v);
}

// e_in = concat([edge_feat, x[src], x[dst]]) as bf16
__global__ void build_ein(const float* __restrict__ edgef, const bf16_t* __restrict__ xb,
                          const int* __restrict__ eidx, bf16_t* __restrict__ ein) {
  int i = blockIdx.x * 256 + threadIdx.x;       // over E*H
  if (i >= kE * kH) return;
  int e = i / kH, h = i - e * kH;
  int s = eidx[e], d = eidx[kE + e];
  bf16_t* o = ein + (size_t)e * (3 * kH);
  o[h]          = f2bf(edgef[i]);
  o[kH + h]     = xb[(size_t)s * kH + h];
  o[2 * kH + h] = xb[(size_t)d * kH + h];
}

// GraphNorm over all edges (dim 0)
__global__ void colmean(const float* __restrict__ e, float* __restrict__ mu) {
  __shared__ float red[256];
  int h = blockIdx.x;
  float s = 0.0f;
  for (int r = threadIdx.x; r < kE; r += 256) s += e[(size_t)r * kH + h];
  float tot = bsum(s, red);
  if (threadIdx.x == 0) mu[h] = tot / (float)kE;
}
__global__ void colvar(const float* __restrict__ e, const float* __restrict__ mu,
                       const float* __restrict__ gn_a, int l, float* __restrict__ var) {
  __shared__ float red[256];
  int h = blockIdx.x;
  float m = gn_a[l * kH + h] * mu[h];
  float s = 0.0f;
  for (int r = threadIdx.x; r < kE; r += 256) {
    float d = e[(size_t)r * kH + h] - m; s += d * d;
  }
  float tot = bsum(s, red);
  if (threadIdx.x == 0) var[h] = tot / (float)kE;
}
__global__ void gnorm_apply(const float* __restrict__ e, const float* __restrict__ mu,
                            const float* __restrict__ var, const float* __restrict__ gn_w,
                            const float* __restrict__ gn_b, const float* __restrict__ gn_a,
                            int l, float* __restrict__ edgef) {
  int i = blockIdx.x * 256 + threadIdx.x;
  if (i >= kE * kH) return;
  int h = i % kH;
  float ec = e[i] - gn_a[l * kH + h] * mu[h];
  edgef[i] = gn_w[l * kH + h] * ec * rsqrtf(var[h] + 1e-5f) + gn_b[l * kH + h];
}

// attention pooling over ALL nodes
__global__ void attn_scores(const float* __restrict__ x, const float* __restrict__ aW,
                            const float* __restrict__ ab, float* __restrict__ sc) {
  __shared__ float red[256];
  int n = blockIdx.x;
  float s = 0.0f;
  for (int h = threadIdx.x; h < kH; h += 256) s += x[(size_t)n * kH + h] * aW[h];
  float tot = bsum(s, red);
  if (threadIdx.x == 0) sc[n] = tot + ab[0];
}
__global__ void node_stats(const float* __restrict__ sc, float* __restrict__ stats) {
  __shared__ float red[1024];
  int t = threadIdx.x;
  float m = -3.4e38f;
  for (int i = t; i < kN; i += 1024) m = fmaxf(m, sc[i]);
  float gm = bmaxr(m, red);
  float s = 0.0f;
  for (int i = t; i < kN; i += 1024) s += expf(sc[i] - gm);
  float gs = bsum(s, red);
  if (t == 0) { stats[0] = gm; stats[1] = gs; }
}
__global__ void pool_nodes(const float* __restrict__ x, const float* __restrict__ sc,
                           const float* __restrict__ stats, float* __restrict__ pooled) {
  int i = blockIdx.x * 256 + threadIdx.x;       // over B*H
  if (i >= kB * kH) return;
  int b = i / kH, h = i - b * kH;
  float gm = stats[0], inv = 1.0f / stats[1];
  float s = 0.0f;
  #pragma unroll
  for (int a = 0; a < kNA; ++a) {
    int n = b * kNA + a;
    s += x[(size_t)n * kH + h] * (expf(sc[n] - gm) * inv);
  }
  pooled[i] = s;
}

// LN(supp)->gelu, combined = pooled + .
__global__ void supp_ln_combine(const float* __restrict__ supp, const float* __restrict__ lnw,
                                const float* __restrict__ lnb, const float* __restrict__ pooled,
                                float* __restrict__ combf, bf16_t* __restrict__ combb) {
  __shared__ float red[256];
  int b = blockIdx.x;
  const float* r = supp + (size_t)b * kH;
  float s = 0.0f;
  for (int h = threadIdx.x; h < kH; h += 256) s += r[h];
  float mean = bsum(s, red) / (float)kH;
  float v = 0.0f;
  for (int h = threadIdx.x; h < kH; h += 256) { float d = r[h] - mean; v += d * d; }
  float var = bsum(v, red) / (float)kH;
  float inv = rsqrtf(var + 1e-5f);
  for (int h = threadIdx.x; h < kH; h += 256) {
    float xn = lnw[h] * (r[h] - mean) * inv + lnb[h];
    float g  = 0.5f * xn * (1.0f + erff(xn * 0.70710678f));
    float c  = pooled[(size_t)b * kH + h] + g;
    combf[(size_t)b * kH + h] = c;
    combb[(size_t)b * kH + h] = f2bf(c);
  }
}

// plain row LayerNorm -> f32 + bf16
__global__ void ln_rows(const float* __restrict__ in, const float* __restrict__ lnw,
                        const float* __restrict__ lnb, float* __restrict__ yf,
                        bf16_t* __restrict__ yb) {
  __shared__ float red[256];
  int b = blockIdx.x;
  const float* r = in + (size_t)b * kH;
  float s = 0.0f;
  for (int h = threadIdx.x; h < kH; h += 256) s += r[h];
  float mean = bsum(s, red) / (float)kH;
  float v = 0.0f;
  for (int h = threadIdx.x; h < kH; h += 256) { float d = r[h] - mean; v += d * d; }
  float var = bsum(v, red) / (float)kH;
  float inv = rsqrtf(var + 1e-5f);
  for (int h = threadIdx.x; h < kH; h += 256) {
    float xn = lnw[h] * (r[h] - mean) * inv + lnb[h];
    yf[(size_t)b * kH + h] = xn;
    yb[(size_t)b * kH + h] = f2bf(xn);
  }
}

// heads: softmax+mask+renorm, mz clamp, interleave [B,512,2]
__global__ void finalize(const float* __restrict__ mzb, const float* __restrict__ lg,
                         float* __restrict__ out) {
  __shared__ float red[256];
  int b = blockIdx.x, t = threadIdx.x;
  const float* L = lg  + (size_t)b * kFrag;
  const float* M = mzb + (size_t)b * kFrag;
  float l0 = L[t], l1 = L[t + 256];
  float gm = bmaxr(fmaxf(l0, l1), red);
  float e0 = expf(l0 - gm), e1 = expf(l1 - gm);
  float gs = bsum(e0 + e1, red);
  float p0 = e0 / gs, p1 = e1 / gs;
  float k0 = (p0 > 1e-4f) ? 1.0f : 0.0f;
  float k1 = (p1 > 1e-4f) ? 1.0f : 0.0f;
  float pm0 = p0 * k0, pm1 = p1 * k1;
  float ps = bsum(pm0 + pm1, red);
  float inv = 1.0f / (ps + 1e-10f);
  float m0 = fminf(fmaxf(M[t], 0.0f), 2000.0f) * k0;
  float m1 = fminf(fmaxf(M[t + 256], 0.0f), 2000.0f) * k1;
  float* ob = out + (size_t)b * kFrag * 2;
  ob[(size_t)t * 2]             = m0;
  ob[(size_t)t * 2 + 1]         = pm0 * inv;
  ob[(size_t)(t + 256) * 2]     = m1;
  ob[(size_t)(t + 256) * 2 + 1] = pm1 * inv;
}

} // namespace

extern "C" void kernel_launch(void* const* d_in, const int* in_sizes, int n_in,
                              void* d_out, int out_size, void* d_ws, size_t ws_size,
                              hipStream_t stream) {
  const float* tok     = (const float*)d_in[0];
  const float* suppin  = (const float*)d_in[1];
  const float* eigvecs = (const float*)d_in[2];
  const float* eigvals = (const float*)d_in[3];
  const int*   eidx    = (const int*)d_in[4];
  const float* phi_W1  = (const float*)d_in[5];
  const float* phi_b1  = (const float*)d_in[6];
  const float* phi_W2  = (const float*)d_in[7];
  const float* phi_b2  = (const float*)d_in[8];
  const float* rho_W1  = (const float*)d_in[9];
  const float* rho_b1  = (const float*)d_in[10];
  const float* rho_W2  = (const float*)d_in[11];
  const float* rho_b2  = (const float*)d_in[12];
  const float* gine_eps= (const float*)d_in[13];
  const float* gine_W1 = (const float*)d_in[14];
  const float* gine_b1 = (const float*)d_in[15];
  const float* gine_W2 = (const float*)d_in[16];
  const float* gine_b2 = (const float*)d_in[17];
  const float* eu_W1   = (const float*)d_in[18];
  const float* eu_b1   = (const float*)d_in[19];
  const float* eu_W2   = (const float*)d_in[20];
  const float* eu_b2   = (const float*)d_in[21];
  const float* gn_w    = (const float*)d_in[22];
  const float* gn_b    = (const float*)d_in[23];
  const float* gn_a    = (const float*)d_in[24];
  const float* attn_W  = (const float*)d_in[25];
  const float* attn_b  = (const float*)d_in[26];
  const float* sup_W   = (const float*)d_in[27];
  const float* sup_b   = (const float*)d_in[28];
  const float* sup_lnw = (const float*)d_in[29];
  const float* sup_lnb = (const float*)d_in[30];
  const float* fc1_W   = (const float*)d_in[31];
  const float* fc1_b   = (const float*)d_in[32];
  const float* fc2_W   = (const float*)d_in[33];
  const float* fc2_b   = (const float*)d_in[34];
  const float* mlp_lnw = (const float*)d_in[35];
  const float* mlp_lnb = (const float*)d_in[36];
  const float* mz_W    = (const float*)d_in[37];
  const float* mz_b    = (const float*)d_in[38];
  const float* prob_W  = (const float*)d_in[39];
  const float* prob_b  = (const float*)d_in[40];

  size_t off = 0;
  auto alloc = [&](size_t bytes) -> void* {
    off = (off + 255) & ~(size_t)255;
    void* p = (char*)d_ws + off;
    off += bytes;
    return p;
  };

  // bf16 weight cache (all stored TRANSPOSED: [N,K])
  bf16_t* wb_rho1 = (bf16_t*)alloc((size_t)256 * kH * 2);
  bf16_t* wb_rho2 = (bf16_t*)alloc((size_t)kH * kH * 2);
  bf16_t* wb_g1   = (bf16_t*)alloc((size_t)kDepth * kH * kH * 2);
  bf16_t* wb_g2   = (bf16_t*)alloc((size_t)kDepth * kH * kH * 2);
  bf16_t* wb_e1   = (bf16_t*)alloc((size_t)kDepth * 3 * kH * kH * 2);
  bf16_t* wb_e2   = (bf16_t*)alloc((size_t)kDepth * kH * kH * 2);
  bf16_t* wb_sup  = (bf16_t*)alloc((size_t)kSupp * kH * 2);
  bf16_t* wb_fc1  = (bf16_t*)alloc((size_t)kH * kH * 2);
  bf16_t* wb_fc2  = (bf16_t*)alloc((size_t)kH * kH * 2);
  bf16_t* wb_mz   = (bf16_t*)alloc((size_t)kH * kFrag * 2);
  bf16_t* wb_prob = (bf16_t*)alloc((size_t)kH * kFrag * 2);
  bf16_t* supp_bf = (bf16_t*)alloc((size_t)kB * kSupp * 2);

  // activations
  float*  nodef = (float*) alloc((size_t)kN * kH * 4);
  float*  edgef = (float*) alloc((size_t)kE * kH * 4);
  bf16_t* hsig  = (bf16_t*)alloc((size_t)kN * kKeig * kPhi * 2);
  bf16_t* tb    = (bf16_t*)alloc((size_t)kN * kH * 2);    // shared bf16 temp
  float*  xf    = (float*) alloc((size_t)kN * kH * 4);
  bf16_t* xb    = (bf16_t*)alloc((size_t)kN * kH * 2);
  float*  scr   = (float*) alloc((size_t)kN * kH * 4);    // agg, then e
  float*  hmf   = (float*) alloc((size_t)kN * kH * 4);
  bf16_t* hmb   = (bf16_t*)alloc((size_t)kN * kH * 2);
  bf16_t* ein   = (bf16_t*)alloc((size_t)kE * 3 * kH * 2);
  float*  mu    = (float*) alloc((size_t)kH * 4);
  float*  var   = (float*) alloc((size_t)kH * 4);
  float*  sc    = (float*) alloc((size_t)kN * 4);
  float*  stats = (float*) alloc(2 * 4);
  float*  pooled= (float*) alloc((size_t)kB * kH * 4);
  float*  suppf = (float*) alloc((size_t)kB * kH * 4);
  float*  combf = (float*) alloc((size_t)kB * kH * 4);
  bf16_t* combb = (bf16_t*)alloc((size_t)kB * kH * 2);
  float*  yresf = (float*) alloc((size_t)kB * kH * 4);
  float*  yf    = (float*) alloc((size_t)kB * kH * 4);
  bf16_t* yb    = (bf16_t*)alloc((size_t)kB * kH * 2);
  float*  mzf   = (float*) alloc((size_t)kB * kFrag * 4);
  float*  lgf   = (float*) alloc((size_t)kB * kFrag * 4);

  auto cvt = [&](const float* s, bf16_t* d, size_t n) {
    cvt_f32_bf16<<<(unsigned)((n + 255) / 256), 256, 0, stream>>>(s, d, (int)n);
  };
  auto cvtT = [&](const float* s, bf16_t* d, int K, int N, int L) {
    dim3 g(K / 32, N / 32, L);
    cvt_transpose<<<g, 256, 0, stream>>>(s, d, K, N);
  };
  auto gemm = [&](const bf16_t* A, const bf16_t* Wt, const float* bias, const float* resid,
                  float* Cf, bf16_t* Cb, int M, int K, int N, int act) {
    dim3 g(M / 128, N / 64);
    gemm_bf16<<<g, 128, 0, stream>>>(A, Wt, bias, resid, Cf, Cb, M, K, N, act);
  };

  // ---- weight conversion + transpose (deterministic, repeated per call) ----
  cvtT(rho_W1, wb_rho1, 256, kH, 1);
  cvtT(rho_W2, wb_rho2, kH, kH, 1);
  cvtT(gine_W1, wb_g1, kH, kH, kDepth);
  cvtT(gine_W2, wb_g2, kH, kH, kDepth);
  cvtT(eu_W1, wb_e1, 3 * kH, kH, kDepth);
  cvtT(eu_W2, wb_e2, kH, kH, kDepth);
  cvtT(sup_W, wb_sup, kSupp, kH, 1);
  cvtT(fc1_W, wb_fc1, kH, kH, 1);
  cvtT(fc2_W, wb_fc2, kH, kH, 1);
  cvtT(mz_W, wb_mz, kH, kFrag, 1);
  cvtT(prob_W, wb_prob, kH, kFrag, 1);
  cvt(suppin, supp_bf, (size_t)kB * kSupp);

  // ---- token -> graph features ----
  build_feats<<<(kB * kH + 255) / 256, 256, 0, stream>>>(tok, nodef, edgef);

  // ---- SignNet ----
  signnet_phi<<<(kN * kKeig + 127) / 128, 128, 0, stream>>>(
      eigvecs, eigvals, phi_W1, phi_b1, phi_W2, phi_b2, hsig);
  gemm(hsig, wb_rho1, rho_b1, nullptr, nullptr, tb, kN, kKeig * kPhi, kH, 1); // relu
  gemm(tb, wb_rho2, rho_b2, nodef, xf, xb, kN, kH, kH, 0);                    // +node_feat

  // ---- GINE layers ----
  for (int l = 0; l < kDepth; ++l) {
    (void)hipMemsetAsync(scr, 0, (size_t)kN * kH * 4, stream);
    gine_agg<<<(kE * kH + 255) / 256, 256, 0, stream>>>(xf, edgef, eidx, scr);
    make_hm<<<(kN * kH + 255) / 256, 256, 0, stream>>>(xf, scr, gine_eps, l, hmf, hmb);
    gemm(hmb, wb_g1 + (size_t)l * kH * kH, gine_b1 + l * kH, nullptr, nullptr, tb,
         kN, kH, kH, 2);                                                       // silu
    gemm(tb, wb_g2 + (size_t)l * kH * kH, gine_b2 + l * kH, hmf, xf, xb,
         kN, kH, kH, 0);                                                       // +hm
    build_ein<<<(kE * kH + 255) / 256, 256, 0, stream>>>(edgef, xb, eidx, ein);
    gemm(ein, wb_e1 + (size_t)l * 3 * kH * kH, eu_b1 + l * kH, nullptr, nullptr, tb,
         kE, 3 * kH, kH, 2);                                                   // silu
    gemm(tb, wb_e2 + (size_t)l * kH * kH, eu_b2 + l * kH, nullptr, scr, nullptr,
         kE, kH, kH, 0);                                                       // e
    colmean<<<kH, 256, 0, stream>>>(scr, mu);
    colvar<<<kH, 256, 0, stream>>>(scr, mu, gn_a, l, var);
    gnorm_apply<<<(kE * kH + 255) / 256, 256, 0, stream>>>(scr, mu, var, gn_w, gn_b,
                                                           gn_a, l, edgef);
  }

  // ---- attention pool ----
  attn_scores<<<kN, 256, 0, stream>>>(xf, attn_W, attn_b, sc);
  node_stats<<<1, 1024, 0, stream>>>(sc, stats);
  pool_nodes<<<(kB * kH + 255) / 256, 256, 0, stream>>>(xf, sc, stats, pooled);

  // ---- supplementary + MLP head ----
  gemm(supp_bf, wb_sup, sup_b, nullptr, suppf, nullptr, kB, kSupp, kH, 0);
  supp_ln_combine<<<kB, 256, 0, stream>>>(suppf, sup_lnw, sup_lnb, pooled, combf, combb);
  gemm(combb, wb_fc1, fc1_b, nullptr, nullptr, tb, kB, kH, kH, 3);             // gelu
  gemm(tb, wb_fc2, fc2_b, combf, yresf, nullptr, kB, kH, kH, 0);               // +combined
  ln_rows<<<kB, 256, 0, stream>>>(yresf, mlp_lnw, mlp_lnb, yf, yb);

  // ---- output heads ----
  gemm(yb, wb_mz,   mz_b,   nullptr, mzf, nullptr, kB, kH, kFrag, 0);
  gemm(yb, wb_prob, prob_b, nullptr, lgf, nullptr, kB, kH, kFrag, 0);
  finalize<<<kB, 256, 0, stream>>>(mzf, lgf, (float*)d_out);

  (void)in_sizes; (void)n_in; (void)out_size; (void)ws_size;
}
